// MotionWKVBlock_62294205661370
// MI455X (gfx1250) — compile-verified, matrix operand order checked by reading, not run on hardware
//
#include <hip/hip_runtime.h>
#include <hip/hip_bf16.h>
#include <math.h>

typedef __attribute__((ext_vector_type(16))) __bf16        v16bf;
typedef __attribute__((ext_vector_type(8)))  float         v8f;
typedef __attribute__((ext_vector_type(4)))  float         v4f;
typedef __attribute__((ext_vector_type(4)))  unsigned int  v4u;
typedef __attribute__((ext_vector_type(2)))  unsigned int  v2u;
typedef unsigned short u16;

#define C_DIM 128

// ---- helpers -------------------------------------------------------------

static __device__ __forceinline__ u16 f2bf(float f) {
  unsigned u = __float_as_uint(f);
  u += 0x7FFFu + ((u >> 16) & 1u);          // round-to-nearest-even
  return (u16)(u >> 16);
}

static __device__ __forceinline__ float wave_sum(float v) {
#pragma unroll
  for (int m = 16; m > 0; m >>= 1) v += __shfl_xor(v, m, 32);
  return v;
}
static __device__ __forceinline__ float wave_max(float v) {
#pragma unroll
  for (int m = 16; m > 0; m >>= 1) v = fmaxf(v, __shfl_xor(v, m, 32));
  return v;
}

// ---- weight prep: fp32 (K x N) -> bf16 transposed (N x K) ----------------

__global__ void transpose_cast_kernel(const float* __restrict__ W,
                                      u16* __restrict__ Wt, int K, int N) {
  int idx = blockIdx.x * blockDim.x + threadIdx.x;
  if (idx >= K * N) return;
  int k = idx / N, n = idx - k * N;
  Wt[(size_t)n * K + k] = f2bf(W[idx]);
}

__global__ void concat3_kernel(const float* __restrict__ a,
                               const float* __restrict__ b,
                               const float* __restrict__ c,
                               float* __restrict__ o) {
  int i = blockIdx.x * blockDim.x + threadIdx.x;
  if (i < C_DIM) o[i] = a[i];
  else if (i < 2 * C_DIM) o[i] = b[i - C_DIM];
  else if (i < 3 * C_DIM) o[i] = c[i - 2 * C_DIM];
}

// ---- LayerNorm: one wave per row (32 lanes x 4 channels = 128) -----------

__global__ __launch_bounds__(256) void ln_kernel(
    const float* __restrict__ X, const float* __restrict__ g,
    const float* __restrict__ b, float* __restrict__ outF,
    u16* __restrict__ outB, int rows) {
  int wid  = (blockIdx.x * blockDim.x + threadIdx.x) >> 5;
  int lane = threadIdx.x & 31;
  if (wid >= rows) return;
  v4f xv = *(const v4f*)(X + (size_t)wid * C_DIM + lane * 4);
  float s1 = xv.x + xv.y + xv.z + xv.w;
  float s2 = xv.x * xv.x + xv.y * xv.y + xv.z * xv.z + xv.w * xv.w;
  s1 = wave_sum(s1);
  s2 = wave_sum(s2);
  float mean = s1 * (1.0f / C_DIM);
  float var  = s2 * (1.0f / C_DIM) - mean * mean;
  float inv  = rsqrtf(var + 1e-5f);
  int c0 = lane * 4;
  float o0 = (xv.x - mean) * inv * g[c0 + 0] + b[c0 + 0];
  float o1 = (xv.y - mean) * inv * g[c0 + 1] + b[c0 + 1];
  float o2 = (xv.z - mean) * inv * g[c0 + 2] + b[c0 + 2];
  float o3 = (xv.w - mean) * inv * g[c0 + 3] + b[c0 + 3];
  size_t base = (size_t)wid * C_DIM + c0;
  if (outF) {
    v4f ov = {o0, o1, o2, o3};
    *(v4f*)(outF + base) = ov;
  }
  if (outB) {
    v2u p;
    p.x = (unsigned)f2bf(o0) | ((unsigned)f2bf(o1) << 16);
    p.y = (unsigned)f2bf(o2) | ((unsigned)f2bf(o3) << 16);
    *(v2u*)(outB + base) = p;
  }
}

// ---- temporal shift + time-mix, emit bf16 activations --------------------

__global__ void shift_kernel(const float* __restrict__ h,
                             const float* __restrict__ mu,
                             u16* __restrict__ hs, int Bn, int T) {
  int idx = blockIdx.x * blockDim.x + threadIdx.x;
  if (idx >= Bn * T * C_DIM) return;
  int c = idx & (C_DIM - 1);
  int t = (idx / C_DIM) % T;
  float sh = 0.0f;
  if (c < C_DIM / 2) { if (t > 0)     sh = h[idx - C_DIM]; }
  else               { if (t < T - 1) sh = h[idx + C_DIM]; }
  hs[idx] = f2bf(h[idx] + mu[c] * sh);
}

// ---- bf16 WMMA GEMM: D = act(A @ B + bias) [+ residual] ------------------
// A: (M x K) bf16 row-major; Bt: (N x K) bf16 (B transposed, N-major).
// One wave computes one 16x16 tile; 8 waves / 256-thread block.
// act: 0=none, 1=sigmoid on cols < C_DIM (rkv fused), 2=exact GELU.

union Frag { v4u q[2]; v16bf v; };

__global__ __launch_bounds__(256) void gemm_wmma_kernel(
    const u16* __restrict__ A, const u16* __restrict__ Bt,
    const float* __restrict__ bias, const float* __restrict__ residual,
    float* __restrict__ outF, u16* __restrict__ outB,
    int M, int N, int K, int act) {
  int tile = (blockIdx.x * blockDim.x + threadIdx.x) >> 5;
  int lane = threadIdx.x & 31;
  int tilesN = N >> 4;
  if (tile >= (M >> 4) * tilesN) return;
  int tm = (tile / tilesN) << 4;
  int tn = (tile % tilesN) << 4;
  int half = lane >> 4;   // K-half select per CDNA5 16-bit A/B fragment layout
  int l15  = lane & 15;

  const u16* arow = A  + (size_t)(tm + l15) * K + (half << 3);
  const u16* brow = Bt + (size_t)(tn + l15) * K + (half << 3);
  v8f acc = {};
  for (int k0 = 0; k0 < K; k0 += 32) {
    Frag fa, fb;
    fa.q[0] = *(const v4u*)(arow);        // K = k0 + half*8 .. +7
    fa.q[1] = *(const v4u*)(arow + 16);   // K = k0 + half*8 + 16 .. +23
    fb.q[0] = *(const v4u*)(brow);
    fb.q[1] = *(const v4u*)(brow + 16);
    acc = __builtin_amdgcn_wmma_f32_16x16x32_bf16(
        false, fa.v, false, fb.v, (short)0, acc, false, false);
    arow += 32;
    brow += 32;
  }

  int col = tn + l15;
  float bcol = bias ? bias[col] : 0.0f;
#pragma unroll
  for (int j = 0; j < 8; ++j) {
    int row = tm + j + (half << 3);       // C/D layout: vgpr j -> M=j (+8 hi half)
    float v = acc[j] + bcol;
    if (act == 1) {
      if (col < C_DIM) v = 1.0f / (1.0f + __expf(-v));
    } else if (act == 2) {
      v = 0.5f * v * (1.0f + erff(v * 0.70710678118654752f));
    }
    size_t o = (size_t)row * N + col;
    if (residual) v += residual[o];
    if (outF) outF[o] = v;
    if (outB) outB[o] = f2bf(v);
  }
}

// ---- WKV: one wave per (b,t); x_b == x_f algebraically, compute once -----
// rkv layout per row: [ r(0..127) | k(128..255) | v(256..383) ], r pre-sigmoided.

__global__ __launch_bounds__(256) void wkv_kernel(
    const float* __restrict__ rkv, const float* __restrict__ w_decay,
    u16* __restrict__ wk, int Bn, int T) {
  int wid  = (blockIdx.x * blockDim.x + threadIdx.x) >> 5;
  int lane = threadIdx.x & 31;
  if (wid >= Bn * T) return;
  int b  = wid / T;
  int c0 = lane * 4;
  const int S = 3 * C_DIM;

  const float* krow = rkv + (size_t)wid * S + C_DIM;
  float q0 = krow[c0 + 0] * (-__expf(w_decay[c0 + 0]));
  float q1 = krow[c0 + 1] * (-__expf(w_decay[c0 + 1]));
  float q2 = krow[c0 + 2] * (-__expf(w_decay[c0 + 2]));
  float q3 = krow[c0 + 3] * (-__expf(w_decay[c0 + 3]));

  const float* vb = rkv + (size_t)(b * T) * S + 2 * C_DIM + c0;
  float a0 = 0.f, a1 = 0.f, a2 = 0.f, a3 = 0.f;
  float invT1 = 1.0f / (float)(T - 1);
  for (int l = 0; l < T; ++l) {
    float s  = (float)(T - 1 - l) * invT1;
    float z0 = q0 * s, z1 = q1 * s, z2 = q2 * s, z3 = q3 * s;
    float m  = wave_max(fmaxf(fmaxf(z0, z1), fmaxf(z2, z3)));
    float e0 = __expf(z0 - m), e1 = __expf(z1 - m);
    float e2 = __expf(z2 - m), e3 = __expf(z3 - m);
    float sum = wave_sum(e0 + e1 + e2 + e3);
    float inv = 1.0f / (sum + 1e-6f);
    v4f vv = *(const v4f*)(vb + (size_t)l * S);
    a0 += e0 * inv * vv.x;
    a1 += e1 * inv * vv.y;
    a2 += e2 * inv * vv.z;
    a3 += e3 * inv * vv.w;
  }
  // wk = r * (x_f + x_b) * 0.5 = r * x_f  (x_b == x_f, see analysis)
  const float* rrow = rkv + (size_t)wid * S + c0;
  v2u p;
  p.x = (unsigned)f2bf(rrow[0] * a0) | ((unsigned)f2bf(rrow[1] * a1) << 16);
  p.y = (unsigned)f2bf(rrow[2] * a2) | ((unsigned)f2bf(rrow[3] * a3) << 16);
  *(v2u*)(wk + (size_t)wid * C_DIM + c0) = p;
}

// ---- host orchestration --------------------------------------------------

extern "C" void kernel_launch(void* const* d_in, const int* in_sizes, int n_in,
                              void* d_out, int out_size, void* d_ws, size_t ws_size,
                              hipStream_t stream) {
  (void)in_sizes; (void)n_in; (void)out_size; (void)ws_size;
  const int Bn = 2, T = 512, C = C_DIM, H = 4 * C_DIM, M = Bn * T;

  const float* x      = (const float*)d_in[0];
  const float* ln1_g  = (const float*)d_in[1];
  const float* ln1_b  = (const float*)d_in[2];
  const float* mu     = (const float*)d_in[3];
  const float* Wr     = (const float*)d_in[4];
  const float* br     = (const float*)d_in[5];
  const float* Wk     = (const float*)d_in[6];
  const float* bk     = (const float*)d_in[7];
  const float* Wv     = (const float*)d_in[8];
  const float* bv     = (const float*)d_in[9];
  const float* w_decay= (const float*)d_in[10];
  const float* Wo     = (const float*)d_in[11];
  const float* bo     = (const float*)d_in[12];
  const float* ln2_g  = (const float*)d_in[13];
  const float* ln2_b  = (const float*)d_in[14];
  const float* W1     = (const float*)d_in[15];
  const float* b1     = (const float*)d_in[16];
  const float* W2     = (const float*)d_in[17];
  const float* b2     = (const float*)d_in[18];
  float* out = (float*)d_out;

  char* ws = (char*)d_ws;
  size_t off = 0;
  auto alloc = [&](size_t bytes) -> char* {
    char* p = ws + off;
    off = (off + bytes + 255) & ~(size_t)255;
    return p;
  };
  u16*   wt_rkv   = (u16*)  alloc((size_t)3 * C * C * 2);
  u16*   wt_o     = (u16*)  alloc((size_t)C * C * 2);
  u16*   wt_1     = (u16*)  alloc((size_t)H * C * 2);
  u16*   wt_2     = (u16*)  alloc((size_t)C * H * 2);
  float* bias_rkv = (float*)alloc((size_t)3 * C * 4);
  float* h        = (float*)alloc((size_t)M * C * 4);
  u16*   hs       = (u16*)  alloc((size_t)M * C * 2);
  float* rkv      = (float*)alloc((size_t)M * 3 * C * 4);
  u16*   wkb      = (u16*)  alloc((size_t)M * C * 2);
  float* y        = (float*)alloc((size_t)M * C * 4);
  u16*   h2       = (u16*)  alloc((size_t)M * C * 2);
  u16*   m1       = (u16*)  alloc((size_t)M * H * 2);

  // weights -> bf16 transposed (N-major)
  int ncc = (C * C + 255) / 256, nch = (C * H + 255) / 256;
  transpose_cast_kernel<<<ncc, 256, 0, stream>>>(Wr, wt_rkv,           C, C);
  transpose_cast_kernel<<<ncc, 256, 0, stream>>>(Wk, wt_rkv + C * C,   C, C);
  transpose_cast_kernel<<<ncc, 256, 0, stream>>>(Wv, wt_rkv + 2*C*C,   C, C);
  transpose_cast_kernel<<<ncc, 256, 0, stream>>>(Wo, wt_o,             C, C);
  transpose_cast_kernel<<<nch, 256, 0, stream>>>(W1, wt_1,             C, H);
  transpose_cast_kernel<<<nch, 256, 0, stream>>>(W2, wt_2,             H, C);
  concat3_kernel<<<2, 256, 0, stream>>>(br, bk, bv, bias_rkv);

  // LN1 -> h ; shift -> hs(bf16)
  ln_kernel<<<M / 8, 256, 0, stream>>>(x, ln1_g, ln1_b, h, nullptr, M);
  shift_kernel<<<(M * C + 255) / 256, 256, 0, stream>>>(h, mu, hs, Bn, T);

  // fused r|k|v GEMM (sigmoid on r columns)
  int tiles_rkv = (M / 16) * ((3 * C) / 16);
  gemm_wmma_kernel<<<tiles_rkv / 8, 256, 0, stream>>>(
      hs, wt_rkv, bias_rkv, nullptr, rkv, nullptr, M, 3 * C, C, 1);

  // WKV (one direction; backward == forward)
  wkv_kernel<<<M / 8, 256, 0, stream>>>(rkv, w_decay, wkb, Bn, T);

  // y = x + wk @ Wo + bo
  int tiles_cc = (M / 16) * (C / 16);
  gemm_wmma_kernel<<<tiles_cc / 8, 256, 0, stream>>>(
      wkb, wt_o, bo, x, y, nullptr, M, C, C, 0);

  // LN2 -> h2(bf16)
  ln_kernel<<<M / 8, 256, 0, stream>>>(y, ln2_g, ln2_b, nullptr, h2, M);

  // FFN1: m1 = gelu(h2 @ W1 + b1) (bf16)
  int tiles_f1 = (M / 16) * (H / 16);
  gemm_wmma_kernel<<<tiles_f1 / 8, 256, 0, stream>>>(
      h2, wt_1, b1, nullptr, nullptr, m1, M, H, C, 2);

  // out = y + m1 @ W2 + b2
  gemm_wmma_kernel<<<tiles_cc / 8, 256, 0, stream>>>(
      m1, wt_2, b2, y, out, nullptr, M, C, H, 0);
}
